// DFFN_81535659147929
// MI455X (gfx1250) — matrix operand order repaired and checked by reading, not compiled, not run on hardware
//
#include <hip/hip_runtime.h>

// ================= CDNA5 WMMA types =================
typedef __attribute__((ext_vector_type(16))) __bf16 v16bf;
typedef __attribute__((ext_vector_type(8)))  float  v8f;

union FragAB { unsigned u[8]; v16bf v; };
union FragC  { float    f[8]; v8f   v; };

// ================= problem constants =================
#define BSZ    4
#define CIN    64
#define C2C    340           // real channels
#define C2P    352           // padded to 22*16 for WMMA M
#define HGT    256
#define WDT    256
#define PPD    32            // patches per spatial dim (256/8)
#define NPAT   4096          // BSZ*PPD*PPD
#define NPIX   64            // 8*8 pixels per patch
#define MIDROW (NPAT*NPIX)   // elements per channel row of mid = 262144

// workspace layout (bytes, 256-aligned)
#define WIN_OFF   0u                       // 352*64  bf16 = 45056
#define WOUT_OFF  45056u                   // 64*192  bf16 = 24576
#define MM_OFF    69632u                   // 340*64*64 bf16 = 2785280
#define MID_OFF   2854912u                 // 352*262144 bf16 = 184549376
// total ~187.4 MB assumed available in d_ws

// ================= helpers =================
__device__ __forceinline__ unsigned short f2bf(float x) {
    unsigned u = __float_as_uint(x);
    unsigned r = u + 0x7FFFu + ((u >> 16) & 1u);   // round-to-nearest-even
    return (unsigned short)(r >> 16);
}
__device__ __forceinline__ float bf2f(unsigned short b) {
    return __uint_as_float(((unsigned)b) << 16);
}

__constant__ float CT8[8] = { 1.0f, 0.70710678118f, 0.0f, -0.70710678118f,
                             -1.0f, -0.70710678118f, 0.0f,  0.70710678118f };

// ================= P1: weight conversion =================
__global__ void k_prep_w(const float* __restrict__ w_in,
                         const float* __restrict__ w_out,
                         unsigned short* __restrict__ winbf,
                         unsigned short* __restrict__ woutbf) {
    int stride = gridDim.x * blockDim.x;
    int tid = blockIdx.x * blockDim.x + threadIdx.x;
    for (int idx = tid; idx < C2P * CIN; idx += stride) {
        int r = idx >> 6;
        winbf[idx] = (r < C2C) ? f2bf(w_in[idx]) : (unsigned short)0;
    }
    for (int idx = tid; idx < 64 * 192; idx += stride) {
        int o = idx / 192, k = idx % 192;
        woutbf[idx] = (k < 170) ? f2bf(w_out[o * 170 + k]) : (unsigned short)0;
    }
}

// ================= P2: build per-channel 64x64 spectral operator =================
// irfft2(F * rfft2(x)) == circular conv with k = irfft2(F_full);
// M_c[o][i] = k_c[(oi-ii)&7][(oj-ij)&7],
// k_c[da][db] = (1/64) * sum_{u,v in 0..7} Ffull[u][v] * cos(2pi(u*da+v*db)/8)
__global__ void k_build_m(const float* __restrict__ fftw,
                          unsigned short* __restrict__ mm) {
    int c = blockIdx.x;
    int t = threadIdx.x;                 // 64 threads: t = da*8+db, later t = o
    __shared__ float kl[64];
    const float* F = fftw + (size_t)c * 40;   // [8][5]
    int da = t >> 3, db = t & 7;
    float s = 0.0f;
    for (int u = 0; u < 8; ++u)
        for (int v = 0; v < 8; ++v) {
            float fv = (v <= 4) ? F[u * 5 + v] : F[((8 - u) & 7) * 5 + (8 - v)];
            s += fv * CT8[(u * da + v * db) & 7];
        }
    kl[t] = s * (1.0f / 64.0f);
    __syncthreads();
    int oi = t >> 3, oj = t & 7;
    unsigned short* row = mm + (size_t)c * 4096 + (size_t)t * 64;
    for (int ii = 0; ii < 8; ++ii)
        for (int ij = 0; ij < 8; ++ij)
            row[ii * 8 + ij] = f2bf(kl[((oi - ii) & 7) * 8 + ((oj - ij) & 7)]);
}

// ================= A: project_in (1x1 conv) via WMMA bf16 =================
// D[c_out 0..351][pix 0..63] per patch; mid[c][patch*64+pix] bf16
__global__ void k_proj_in(const float* __restrict__ x,
                          const unsigned short* __restrict__ winbf,
                          unsigned short* __restrict__ mid) {
    __shared__ unsigned short WinL[C2P * 66];   // [m 352][k 64], stride 66 (anti-bank-conflict)
    __shared__ unsigned short XL[64 * 66];      // [pix 64][c 64], stride 66

    int patch = blockIdx.x;
    int b  = patch >> 10;
    int pr = (patch >> 5) & 31;
    int pc = patch & 31;
    int tid = threadIdx.x;

    // stage W_in (bf16 pairs as dwords)
    const unsigned* ws = (const unsigned*)winbf;
    for (int d = tid; d < C2P * 32; d += 256) {
        int row = d >> 5, cd = d & 31;
        *(unsigned*)&WinL[row * 66 + 2 * cd] = ws[d];
    }
    // stage X patch (fp32 -> bf16, transposed to [pix][c])
    const float* xb = x + (size_t)b * CIN * (HGT * WDT);
    for (int r = 0; r < 2; ++r) {
        int idx = tid + 256 * r;                 // 512 (c,i) row jobs
        int c = idx >> 3, i = idx & 7;
        const float* src = xb + (size_t)c * (HGT * WDT)
                              + (size_t)(pr * 8 + i) * WDT + pc * 8;
        #pragma unroll
        for (int j = 0; j < 8; ++j)
            XL[(i * 8 + j) * 66 + c] = f2bf(src[j]);
    }
    __syncthreads();

    int lane = tid & 31, wid = tid >> 5;
    int ln = lane & 15, hh = lane >> 4;

    for (int t = wid; t < 88; t += 8) {          // 22 M-tiles x 4 N-tiles
        int tm = t >> 2, tn = t & 3;
        FragC acc;
        #pragma unroll
        for (int v = 0; v < 8; ++v) acc.f[v] = 0.0f;

        #pragma unroll
        for (int kk = 0; kk < 64; kk += 32) {
            FragAB a, bb;
            #pragma unroll
            for (int v = 0; v < 8; ++v) {
                int ka = kk + ((v & 4) ? 16 : 0) + hh * 8 + (v & 3) * 2;
                a.u[v]  = *(const unsigned*)&WinL[(tm * 16 + ln) * 66 + ka];
                int kb = kk + ((v & 4) ? 16 : 0) + hh * 8 + (v & 3) * 2;
                bb.u[v] = *(const unsigned*)&XL[(tn * 16 + ln) * 66 + kb];
            }
            acc.v = __builtin_amdgcn_wmma_f32_16x16x32_bf16(
                        false, a.v, false, bb.v, (short)0, acc.v, false, false);
        }
        size_t baseP = (size_t)patch * 64 + (size_t)(tn * 16 + ln);
        #pragma unroll
        for (int v = 0; v < 8; ++v) {
            int row = tm * 16 + v + 8 * hh;
            mid[(size_t)row * MIDROW + baseP] = f2bf(acc.f[v]);
        }
    }
}

// ================= F: spectral mix (per-channel 64x64 op) via WMMA, in place =================
__global__ void k_specmix(const unsigned short* __restrict__ mm,
                          unsigned short* __restrict__ mid) {
    __shared__ unsigned short McL[64 * 66];   // [o 64][i 64] stride 66
    __shared__ unsigned short inT[32 * 66];   // [patch 32][pix 64] stride 66
    __shared__ unsigned short outT[32 * 66];

    int c  = blockIdx.x >> 7;                 // 340 channels
    int pb = blockIdx.x & 127;                // 128 patch-blocks of 32
    int p0 = pb * 32;
    int tid = threadIdx.x;

    const unsigned* ms = (const unsigned*)(mm + (size_t)c * 4096);
    for (int d = tid; d < 2048; d += 256) {
        int row = d >> 5, cd = d & 31;
        *(unsigned*)&McL[row * 66 + 2 * cd] = ms[d];
    }
    const unsigned* is = (const unsigned*)(mid + (size_t)c * MIDROW + (size_t)p0 * 64);
    for (int d = tid; d < 1024; d += 256) {
        int row = d >> 5, cd = d & 31;
        *(unsigned*)&inT[row * 66 + 2 * cd] = is[d];
    }
    __syncthreads();

    int lane = tid & 31, wid = tid >> 5;
    int ln = lane & 15, hh = lane >> 4;
    int tm = wid >> 1, tn = wid & 1;          // 4 M-tiles x 2 N-tiles = 8 waves

    FragC acc;
    #pragma unroll
    for (int v = 0; v < 8; ++v) acc.f[v] = 0.0f;
    #pragma unroll
    for (int kk = 0; kk < 64; kk += 32) {
        FragAB a, bb;
        #pragma unroll
        for (int v = 0; v < 8; ++v) {
            int ka = kk + ((v & 4) ? 16 : 0) + hh * 8 + (v & 3) * 2;
            a.u[v]  = *(const unsigned*)&McL[(tm * 16 + ln) * 66 + ka];
            int kb = kk + ((v & 4) ? 16 : 0) + hh * 8 + (v & 3) * 2;
            bb.u[v] = *(const unsigned*)&inT[(tn * 16 + ln) * 66 + kb];
        }
        acc.v = __builtin_amdgcn_wmma_f32_16x16x32_bf16(
                    false, a.v, false, bb.v, (short)0, acc.v, false, false);
    }
    // transpose-stage D (rows = out-pixel) into [patch][pix]
    #pragma unroll
    for (int v = 0; v < 8; ++v) {
        int opix = tm * 16 + v + 8 * hh;
        outT[(tn * 16 + ln) * 66 + opix] = f2bf(acc.f[v]);
    }
    __syncthreads();

    unsigned* od = (unsigned*)(mid + (size_t)c * MIDROW + (size_t)p0 * 64);
    for (int d = tid; d < 1024; d += 256) {
        int row = d >> 5, cd = d & 31;
        od[d] = *(const unsigned*)&outT[row * 66 + 2 * cd];
    }
}

// ================= B: depthwise 3x3 + GELU gate + project_out WMMA =================
__device__ __forceinline__ float dwconv(const unsigned short* __restrict__ mid,
                                        const float* __restrict__ wdw,
                                        int ch, int b, int pr, int pc, int i, int j) {
    int h = pr * 8 + i, w = pc * 8 + j;
    float acc = 0.0f;
    #pragma unroll
    for (int ky = 0; ky < 3; ++ky) {
        int hh = h + ky - 1;
        if ((unsigned)hh < (unsigned)HGT) {
            int pr2 = hh >> 3, i2 = hh & 7;
            #pragma unroll
            for (int kx = 0; kx < 3; ++kx) {
                int ww = w + kx - 1;
                if ((unsigned)ww < (unsigned)WDT) {
                    int pc2 = ww >> 3, j2 = ww & 7;
                    size_t idx = (size_t)ch * MIDROW
                               + (size_t)(((b * PPD + pr2) * PPD + pc2)) * 64
                               + (size_t)(i2 * 8 + j2);
                    acc += wdw[ch * 9 + ky * 3 + kx] * bf2f(mid[idx]);
                }
            }
        }
    }
    return acc;
}

__global__ void k_dw_gate_out(const unsigned short* __restrict__ mid,
                              const float* __restrict__ wdw,
                              const unsigned short* __restrict__ woutbf,
                              float* __restrict__ out) {
    __shared__ unsigned short WoL[64 * 194];  // [o 64][k 192] stride 194
    __shared__ unsigned short GL[64 * 194];   // [pix 64][k 192] stride 194

    int patch = blockIdx.x;
    int b  = patch >> 10;
    int pr = (patch >> 5) & 31;
    int pc = patch & 31;
    int tid = threadIdx.x;

    const unsigned* wsrc = (const unsigned*)woutbf;      // 64*96 dwords
    for (int d = tid; d < 64 * 96; d += 256) {
        int row = d / 96, cd = d % 96;
        *(unsigned*)&WoL[row * 194 + 2 * cd] = wsrc[d];
    }
    // depthwise conv + exact-GELU gate -> GL[pix][c] bf16
    for (int idx = tid; idx < 170 * 64; idx += 256) {
        int ch = idx >> 6, pix = idx & 63;
        int i = pix >> 3, j = pix & 7;
        float d1 = dwconv(mid, wdw, ch,       b, pr, pc, i, j);
        float d2 = dwconv(mid, wdw, ch + 170, b, pr, pc, i, j);
        float g = 0.5f * d1 * (1.0f + erff(d1 * 0.70710678f)) * d2;
        GL[pix * 194 + ch] = f2bf(g);
    }
    for (int idx = tid; idx < 64 * 22; idx += 256) {      // zero K pad 170..191
        int pix = idx / 22, jj = idx % 22;
        GL[pix * 194 + 170 + jj] = 0;
    }
    __syncthreads();

    int lane = tid & 31, wid = tid >> 5;
    int ln = lane & 15, hh = lane >> 4;

    for (int t = wid; t < 16; t += 8) {       // 4 M-tiles x 4 N-tiles
        int tm = t >> 2, tn = t & 3;
        FragC acc;
        #pragma unroll
        for (int v = 0; v < 8; ++v) acc.f[v] = 0.0f;
        #pragma unroll
        for (int kk = 0; kk < 192; kk += 32) {
            FragAB a, bb;
            #pragma unroll
            for (int v = 0; v < 8; ++v) {
                int ka = kk + ((v & 4) ? 16 : 0) + hh * 8 + (v & 3) * 2;
                a.u[v]  = *(const unsigned*)&WoL[(tm * 16 + ln) * 194 + ka];
                int kb = kk + ((v & 4) ? 16 : 0) + hh * 8 + (v & 3) * 2;
                bb.u[v] = *(const unsigned*)&GL[(tn * 16 + ln) * 194 + kb];
            }
            acc.v = __builtin_amdgcn_wmma_f32_16x16x32_bf16(
                        false, a.v, false, bb.v, (short)0, acc.v, false, false);
        }
        #pragma unroll
        for (int v = 0; v < 8; ++v) {
            int orow = tm * 16 + v + 8 * hh;
            int pix  = tn * 16 + ln;
            size_t oidx = (((size_t)b * 64 + orow) * HGT + (size_t)(pr * 8 + (pix >> 3))) * WDT
                        + (size_t)(pc * 8 + (pix & 7));
            out[oidx] = acc.f[v];
        }
    }
}

// ================= host launch =================
extern "C" void kernel_launch(void* const* d_in, const int* in_sizes, int n_in,
                              void* d_out, int out_size, void* d_ws, size_t ws_size,
                              hipStream_t stream) {
    const float* x     = (const float*)d_in[0];
    const float* w_in  = (const float*)d_in[1];
    const float* w_dw  = (const float*)d_in[2];
    const float* fft_w = (const float*)d_in[3];
    const float* w_out = (const float*)d_in[4];
    float* out = (float*)d_out;

    unsigned char* ws = (unsigned char*)d_ws;
    unsigned short* winbf  = (unsigned short*)(ws + WIN_OFF);
    unsigned short* woutbf = (unsigned short*)(ws + WOUT_OFF);
    unsigned short* mm     = (unsigned short*)(ws + MM_OFF);
    unsigned short* mid    = (unsigned short*)(ws + MID_OFF);

    k_prep_w<<<64, 256, 0, stream>>>(w_in, w_out, winbf, woutbf);
    k_build_m<<<C2C, 64, 0, stream>>>(fft_w, mm);
    k_proj_in<<<NPAT, 256, 0, stream>>>(x, winbf, mid);
    k_specmix<<<C2C * 128, 256, 0, stream>>>(mm, mid);
    k_dw_gate_out<<<NPAT, 256, 0, stream>>>(mid, w_dw, woutbf, out);
}
